// EvoformerStack_3358664425718
// MI455X (gfx1250) — matrix operand order repaired and checked by reading, not compiled, not run on hardware
//
#include <hip/hip_runtime.h>

// ============================================================================
// Evoformer (2 blocks) + EGNN head for MI455X (gfx1250, wave32, WMMA).
// All dense matmuls run through v_wmma_f32_16x16x32_f16 (f16 in, f32 acc).
// GEMM: one wave computes a 32x32 macro-tile (2x2 WMMA subtiles); branch-free
// clamped loads on edges; vectorized fragment loads when strides allow.
// Workspace requirement: ~259 MB (64,782,336 floats).
// ============================================================================

typedef __attribute__((ext_vector_type(16))) _Float16 v16h;
typedef __attribute__((ext_vector_type(8)))  float    v8f;

__device__ __forceinline__ v8f wmma32(const v16h& a, const v16h& b, const v8f& c) {
  return __builtin_amdgcn_wmma_f32_16x16x32_f16(false, a, false, b, (short)0, c, false, false);
}

// k-offset of A-fragment half h (per-lane kaHi folded into base).
// Contiguous K order (matches B fragment).
__device__ __forceinline__ int akoff(int h) {
  return h;
}

template <bool AV>
__device__ __forceinline__ v16h load_a(const float* __restrict__ base, int k0, int csA) {
  v16h a;
  if (AV) {  // csA==1: adjacent K pairs -> float2 loads
#pragma unroll
    for (int p = 0; p < 8; ++p) {
      const int off = k0 + 2 * p;
      const float2 f = *(const float2*)(base + off);
      a[2 * p]     = (_Float16)f.x;
      a[2 * p + 1] = (_Float16)f.y;
    }
  } else {
#pragma unroll
    for (int h = 0; h < 16; ++h)
      a[h] = (_Float16)base[(k0 + akoff(h)) * csA];
  }
  return a;
}

template <bool BV>
__device__ __forceinline__ v16h load_b(const float* __restrict__ base, int k0, int rsB) {
  v16h b;
  if (BV) {  // rsB==1: 16 consecutive K -> four float4 loads
#pragma unroll
    for (int q = 0; q < 4; ++q) {
      const float4 f = *(const float4*)(base + k0 + 4 * q);
      b[4 * q + 0] = (_Float16)f.x;
      b[4 * q + 1] = (_Float16)f.y;
      b[4 * q + 2] = (_Float16)f.z;
      b[4 * q + 3] = (_Float16)f.w;
    }
  } else {
#pragma unroll
    for (int h = 0; h < 16; ++h)
      b[h] = (_Float16)base[(k0 + h) * rsB];
  }
  return b;
}

// Guarded loads: clamped address + value select (no EXEC branches).
__device__ __forceinline__ v16h load_a_g(const float* __restrict__ base, int k0,
                                         int csA, int kLim) {
  v16h a;
#pragma unroll
  for (int h = 0; h < 16; ++h) {
    const int k = k0 + akoff(h);
    const bool ok = k < kLim;
    const float x = base[ok ? k * csA : 0];
    a[h] = ok ? (_Float16)x : (_Float16)0.f;
  }
  return a;
}
__device__ __forceinline__ v16h load_b_g(const float* __restrict__ base, int k0,
                                         int rsB, int kLim) {
  v16h b;
#pragma unroll
  for (int h = 0; h < 16; ++h) {
    const int k = k0 + h;
    const bool ok = k < kLim;
    const float x = base[ok ? k * rsB : 0];
    b[h] = ok ? (_Float16)x : (_Float16)0.f;
  }
  return b;
}

template <bool AV, bool BV>
__device__ __forceinline__ void gemm_core(const float* __restrict__ a0,
                                          const float* __restrict__ a1,
                                          const float* __restrict__ b0,
                                          const float* __restrict__ b1,
                                          int csA, int rsB, int kFull,
                                          v8f& c00, v8f& c01, v8f& c10, v8f& c11) {
  for (int k0 = 0; k0 < kFull; k0 += 32) {
    const v16h A0 = load_a<AV>(a0, k0, csA);
    const v16h A1 = load_a<AV>(a1, k0, csA);
    const v16h B0 = load_b<BV>(b0, k0, rsB);
    const v16h B1 = load_b<BV>(b1, k0, rsB);
    c00 = wmma32(A0, B0, c00);
    c01 = wmma32(A0, B1, c01);
    c10 = wmma32(A1, B0, c10);
    c11 = wmma32(A1, B1, c11);
  }
}

__device__ __forceinline__ void epi(float* __restrict__ Ob, const v8f& acc,
                                    int tm16, int tn16, int lane, int rsO, int csO,
                                    int M, int N, const float* __restrict__ bias,
                                    float scale, int act, int accum) {
  const int nO = tn16 + (lane & 15);
  const int mB = tm16 + ((lane & 16) ? 8 : 0);
  float bv = 0.f;
  if (bias && nO < N) bv = bias[nO];
#pragma unroll
  for (int r = 0; r < 8; ++r) {
    const int m = mB + r;
    if (m < M && nO < N) {
      float x = acc[r] * scale + bv;
      if      (act == 1) x = fmaxf(x, 0.f);
      else if (act == 2) x = 1.f / (1.f + __expf(-x));
      else if (act == 3) x = x / (1.f + __expf(-x));
      const long long oi = (long long)m * rsO + (long long)nO * csO;
      if (accum) Ob[oi] += x; else Ob[oi] = x;
    }
  }
}

// ---------------------------------------------------------------------------
// Generic strided, batched GEMM:  O = act(scale * (A x B) + bias) [+= if accum]
// 128 threads = 4 waves; wave w handles M-macro-tile blockIdx.y*4+w (32 rows),
// 32 columns per block. batch b = b2*bDiv + b1 -> offsets b1*bs?1 + b2*bs?2.
// act: 0 none, 1 relu, 2 sigmoid, 3 silu
// ---------------------------------------------------------------------------
__global__ void __launch_bounds__(128)
k_gemm(const float* __restrict__ A, long long offA, int rsA, int csA,
       long long bsA1, long long bsA2,
       const float* __restrict__ B, long long offB, int rsB, int csB,
       long long bsB1, long long bsB2,
       const float* __restrict__ bias,
       float* __restrict__ O, long long offO, int rsO, int csO,
       long long bsO1, long long bsO2,
       int M, int N, int K, int bDiv, float scale, int act, int accum)
{
  const int lane = threadIdx.x & 31;
  const int wv   = threadIdx.x >> 5;
  const int tmT  = blockIdx.y * 4 + wv;
  if (tmT * 32 >= M) return;
  const int tm = tmT * 32;
  const int tn = blockIdx.x * 32;
  const int bb1 = blockIdx.z % bDiv;
  const int bb2 = blockIdx.z / bDiv;

  const float* __restrict__ Ab = A + offA + bb1 * bsA1 + bb2 * bsA2;
  const float* __restrict__ Bb = B + offB + bb1 * bsB1 + bb2 * bsB2;
  float*       __restrict__ Ob = O + offO + bb1 * bsO1 + bb2 * bsO2;

  const int lr   = lane & 15;
  const int kaHi = (lane & 16) ? 16 : 0;
  const int kbHi = (lane & 16) ? 16 : 0;
  const int kFull = K & ~31;

  if (tm + 32 <= M && tn + 32 <= N) {
    // ---------------- fast path: full 32x32 macro-tile ----------------
    const float* a0  = Ab + (long long)(tm + lr) * rsA + (long long)kaHi * csA;
    const float* a1  = a0 + 16LL * rsA;
    const float* b0  = Bb + (long long)(tn + lr) * csB + (long long)kbHi * rsB;
    const float* b1p = b0 + 16LL * csB;
    v8f c00 = {0.f,0.f,0.f,0.f,0.f,0.f,0.f,0.f};
    v8f c01 = c00, c10 = c00, c11 = c00;

    const bool av = (csA == 1) && ((rsA & 1) == 0) &&
                    ((((unsigned long long)Ab) & 7) == 0);
    const bool bvec = (rsB == 1) && ((csB & 3) == 0) &&
                      ((((unsigned long long)Bb) & 15) == 0);
    if (av) {
      if (bvec) gemm_core<true,  true >(a0, a1, b0, b1p, csA, rsB, kFull, c00, c01, c10, c11);
      else      gemm_core<true,  false>(a0, a1, b0, b1p, csA, rsB, kFull, c00, c01, c10, c11);
    } else {
      if (bvec) gemm_core<false, true >(a0, a1, b0, b1p, csA, rsB, kFull, c00, c01, c10, c11);
      else      gemm_core<false, false>(a0, a1, b0, b1p, csA, rsB, kFull, c00, c01, c10, c11);
    }
    if (kFull < K) {  // K tail (K not multiple of 32)
      const v16h A0 = load_a_g(a0,  kFull, csA, K - kaHi);
      const v16h A1 = load_a_g(a1,  kFull, csA, K - kaHi);
      const v16h B0 = load_b_g(b0,  kFull, rsB, K - kbHi);
      const v16h B1 = load_b_g(b1p, kFull, rsB, K - kbHi);
      c00 = wmma32(A0, B0, c00);
      c01 = wmma32(A0, B1, c01);
      c10 = wmma32(A1, B0, c10);
      c11 = wmma32(A1, B1, c11);
    }
    epi(Ob, c00, tm,      tn,      lane, rsO, csO, M, N, bias, scale, act, accum);
    epi(Ob, c01, tm,      tn + 16, lane, rsO, csO, M, N, bias, scale, act, accum);
    epi(Ob, c10, tm + 16, tn,      lane, rsO, csO, M, N, bias, scale, act, accum);
    epi(Ob, c11, tm + 16, tn + 16, lane, rsO, csO, M, N, bias, scale, act, accum);
  } else {
    // ---------------- guarded path: 16x16 subtiles, clamped loads ----------
    for (int sm = 0; sm < 2; ++sm) {
      const int tmm = tm + sm * 16;
      if (tmm >= M) break;
      const int mA   = tmm + lr;
      const int mC   = (mA < M) ? mA : (M - 1);
      const int kLA  = (mA < M) ? (K - kaHi) : 0;
      const float* a0 = Ab + (long long)mC * rsA + (long long)kaHi * csA;
      for (int sn = 0; sn < 2; ++sn) {
        const int tnn = tn + sn * 16;
        if (tnn >= N) break;
        const int nB   = tnn + lr;
        const int nC   = (nB < N) ? nB : (N - 1);
        const int kLB  = (nB < N) ? (K - kbHi) : 0;
        const float* b0 = Bb + (long long)nC * csB + (long long)kbHi * rsB;
        v8f c = {0.f,0.f,0.f,0.f,0.f,0.f,0.f,0.f};
        for (int k0 = 0; k0 < K; k0 += 32) {
          const v16h A0 = load_a_g(a0, k0, csA, kLA);
          const v16h B0 = load_b_g(b0, k0, rsB, kLB);
          c = wmma32(A0, B0, c);
        }
        epi(Ob, c, tmm, tnn, lane, rsO, csO, M, N, bias, scale, act, accum);
      }
    }
  }
}

// ---------------------------------------------------------------------------
// LayerNorm over last dim C; one wave per row.
// ---------------------------------------------------------------------------
__global__ void k_layernorm(const float* __restrict__ x, const float* __restrict__ g,
                            const float* __restrict__ b, float* __restrict__ o,
                            int rows, int C)
{
  const int wpb  = blockDim.x >> 5;
  const int wid  = blockIdx.x * wpb + (threadIdx.x >> 5);
  const int lane = threadIdx.x & 31;
  if (wid >= rows) return;
  const float* xr = x + (long long)wid * C;
  float s = 0.f, s2 = 0.f;
  for (int c = lane; c < C; c += 32) { float v = xr[c]; s += v; s2 += v * v; }
#pragma unroll
  for (int off = 16; off > 0; off >>= 1) {
    s  += __shfl_xor(s,  off, 32);
    s2 += __shfl_xor(s2, off, 32);
  }
  const float mu  = s / C;
  const float var = s2 / C - mu * mu;
  const float rsd = rsqrtf(var + 1e-5f);
  float* orow = o + (long long)wid * C;
  for (int c = lane; c < C; c += 32) orow[c] = (xr[c] - mu) * rsd * g[c] + b[c];
}

// ---------------------------------------------------------------------------
// Softmax over rows of length L; one wave per row.
// ---------------------------------------------------------------------------
__global__ void k_softmax(float* __restrict__ x, long long rows, int L)
{
  const int wpb = blockDim.x >> 5;
  const long long wid = (long long)blockIdx.x * wpb + (threadIdx.x >> 5);
  const int lane = threadIdx.x & 31;
  if (wid >= rows) return;
  float* r = x + wid * L;
  float mx = -3.4e38f;
  for (int i = lane; i < L; i += 32) mx = fmaxf(mx, r[i]);
#pragma unroll
  for (int off = 16; off > 0; off >>= 1) mx = fmaxf(mx, __shfl_xor(mx, off, 32));
  float s = 0.f;
  for (int i = lane; i < L; i += 32) { float e = __expf(r[i] - mx); r[i] = e; s += e; }
#pragma unroll
  for (int off = 16; off > 0; off >>= 1) s += __shfl_xor(s, off, 32);
  const float inv = 1.f / s;
  for (int i = lane; i < L; i += 32) r[i] *= inv;
}

// ---------------- elementwise helpers ----------------
__global__ void k_mul(float* __restrict__ o, const float* __restrict__ a,
                      const float* __restrict__ b, long long n)
{
  long long i = (long long)blockIdx.x * blockDim.x + threadIdx.x;
  if (i < n) o[i] = a[i] * b[i];
}

__global__ void k_fma(float* __restrict__ d, const float* __restrict__ a,
                      const float* __restrict__ b, long long n)
{
  long long i = (long long)blockIdx.x * blockDim.x + threadIdx.x;
  if (i < n) d[i] += a[i] * b[i];
}

// logits[s,h,i,j] += bias[(i*R+j)*H + h]
__global__ void k_addbias_row(float* __restrict__ lg, const float* __restrict__ bias,
                              int S, int H, int R)
{
  long long idx = (long long)blockIdx.x * blockDim.x + threadIdx.x;
  long long n = (long long)S * H * R * R;
  if (idx >= n) return;
  int j = (int)(idx % R);
  int i = (int)((idx / R) % R);
  int h = (int)((idx / ((long long)R * R)) % H);
  lg[idx] += bias[((long long)i * R + j) * H + h];
}

// logits[i,j,k] += bias[(j*R+k)*H + h]   (single head h)
__global__ void k_addbias_tri(float* __restrict__ lg, const float* __restrict__ bias,
                              int h, int H, int R)
{
  long long idx = (long long)blockIdx.x * blockDim.x + threadIdx.x;
  long long n = (long long)R * R * R;
  if (idx >= n) return;
  int k = (int)(idx % R);
  int j = (int)((idx / R) % R);
  lg[idx] += bias[((long long)j * R + k) * H + h];
}

// dst[i,j,c] = src[j,i,c]
__global__ void k_transpose(float* __restrict__ dst, const float* __restrict__ src,
                            int R, int C)
{
  long long idx = (long long)blockIdx.x * blockDim.x + threadIdx.x;
  long long n = (long long)R * R * C;
  if (idx >= n) return;
  int c = (int)(idx % C);
  long long rem = idx / C;
  int j = (int)(rem % R);
  int i = (int)(rem / R);
  dst[idx] = src[((long long)j * R + i) * C + c];
}

// z[i,j,c] += src[j,i,c]
__global__ void k_add_t(float* __restrict__ z, const float* __restrict__ src,
                        int R, int C)
{
  long long idx = (long long)blockIdx.x * blockDim.x + threadIdx.x;
  long long n = (long long)R * R * C;
  if (idx >= n) return;
  int c = (int)(idx % C);
  long long rem = idx / C;
  int j = (int)(rem % R);
  int i = (int)(rem / R);
  z[idx] += src[((long long)j * R + i) * C + c];
}

// ---------------------------------------------------------------------------
// Outer product mean, fused: for block (i,j):
//   M[c,d] = (1/S) sum_s a[s,i,c] * b[s,j,d];  z[i,j,e] += M_flat . Wo[:,e] + bo[e]
// grid (R, R), block 128 threads.
// ---------------------------------------------------------------------------
__global__ void k_opm(const float* __restrict__ a, const float* __restrict__ bm,
                      const float* __restrict__ Wo, const float* __restrict__ bo,
                      float* __restrict__ z, int S, int R)
{
  __shared__ float sa[64 * 32];
  __shared__ float sb[64 * 32];
  __shared__ float sm[32 * 32];
  const int j = blockIdx.x, i = blockIdx.y;
  const int t = threadIdx.x;           // 128 threads
  for (int idx = t; idx < S * 32; idx += 128) {
    const int s = idx >> 5, c = idx & 31;
    sa[idx] = a [((long long)s * R + i) * 32 + c];
    sb[idx] = bm[((long long)s * R + j) * 32 + c];
  }
  __syncthreads();
  const float inv = 1.f / (float)S;
  for (int e = t; e < 1024; e += 128) {
    const int c = e >> 5, d = e & 31;
    float acc = 0.f;
    for (int s = 0; s < S; ++s) acc += sa[s * 32 + c] * sb[s * 32 + d];
    sm[e] = acc * inv;
  }
  __syncthreads();
  float acc = bo[t];
  for (int cd = 0; cd < 1024; ++cd) acc += sm[cd] * Wo[cd * 128 + t];
  z[((long long)i * R + j) * 128 + t] += acc;
}

// ---------------- EGNN glue ----------------
__global__ void k_egnn_edges(const float* __restrict__ s, const float* __restrict__ co,
                             float* __restrict__ ein, int N)
{
  long long idx = (long long)blockIdx.x * blockDim.x + threadIdx.x;
  const int E = 2 * (N - 1), F = 769;
  if (idx >= (long long)E * F) return;
  const int f = (int)(idx % F);
  const int e = (int)(idx / F);
  const int r = (e < N - 1) ? e : e - (N - 2);
  const int c = (e < N - 1) ? e + 1 : e - (N - 1);
  float v;
  if (f < 384)       v = s[r * 384 + f];
  else if (f < 768)  v = s[c * 384 + (f - 384)];
  else {
    const float dx = co[r * 3 + 0] - co[c * 3 + 0];
    const float dy = co[r * 3 + 1] - co[c * 3 + 1];
    const float dz = co[r * 3 + 2] - co[c * 3 + 2];
    v = dx * dx + dy * dy + dz * dz;
  }
  ein[idx] = v;
}

__global__ void k_egnn_agg(const float* __restrict__ ef, float* __restrict__ agg, int N)
{
  long long idx = (long long)blockIdx.x * blockDim.x + threadIdx.x;
  if (idx >= (long long)N * 384) return;
  const int f = (int)(idx % 384);
  const int n = (int)(idx / 384);
  float v = 0.f;
  if (n <= N - 2) v += ef[n * 384 + f];
  if (n >= 1)     v += ef[(n + N - 2) * 384 + f];
  agg[idx] = v;
}

__global__ void k_concat2(float* __restrict__ dst, const float* __restrict__ a,
                          const float* __restrict__ b, int N, int F)
{
  long long idx = (long long)blockIdx.x * blockDim.x + threadIdx.x;
  if (idx >= (long long)N * 2 * F) return;
  const int f = (int)(idx % (2 * F));
  const int n = (int)(idx / (2 * F));
  dst[idx] = (f < F) ? a[n * F + f] : b[n * F + f - F];
}

// ============================================================================
// Host side
// ============================================================================

struct BlockP {
  const float *ca_Wg,*ca_Wk,*ca_Wo,*ca_Wq,*ca_Wv,*ca_bg,*ca_bo,*ca_lnb,*ca_lng;
  const float *mt_W1,*mt_W2,*mt_b1,*mt_b2,*mt_lnb,*mt_lng;
  const float *op_Wa,*op_Wb,*op_Wo,*op_ba,*op_bb,*op_bo,*op_lnb,*op_lng;
  const float *pt_W1,*pt_W2,*pt_b1,*pt_b2,*pt_lnb,*pt_lng;
  const float *ra_Wb2,*ra_Wg,*ra_Wk,*ra_Wo,*ra_Wq,*ra_Wv,*ra_bg,*ra_bo,*ra_lnb,*ra_lng,*ra_lnzb,*ra_lnzg;
  const float *te_Wb2,*te_Wg,*te_Wk,*te_Wo,*te_Wq,*te_Wv,*te_bg,*te_bo,*te_lnb,*te_lng;
  const float *ts_Wb2,*ts_Wg,*ts_Wk,*ts_Wo,*ts_Wq,*ts_Wv,*ts_bg,*ts_bo,*ts_lnb,*ts_lng;
  const float *ti_Wa,*ti_Wag,*ti_Wb,*ti_Wbg,*ti_Wg,*ti_Wo,*ti_ba,*ti_bag,*ti_bb,*ti_bbg,*ti_bg,*ti_bo,*ti_ln2b,*ti_ln2g,*ti_lnb,*ti_lng;
  const float *to_Wa,*to_Wag,*to_Wb,*to_Wbg,*to_Wg,*to_Wo,*to_ba,*to_bag,*to_bb,*to_bbg,*to_bg,*to_bo,*to_ln2b,*to_ln2g,*to_lnb,*to_lng;
};

extern "C" void kernel_launch(void* const* d_in, const int* in_sizes, int n_in,
                              void* d_out, int out_size, void* d_ws, size_t ws_size,
                              hipStream_t stream)
{
  (void)in_sizes; (void)out_size; (void)ws_size;
  const int S = 64, R = 192;
  const long long M_ELT = 3145728;   // 1*64*192*256
  const long long Z_ELT = 4718592;   // 1*192*192*128
  const float rs32 = 0.17677669529663687f;  // 1/sqrt(32)

  // ---------------- parse inputs (JAX pytree order: sorted dict keys) -------
  const float* m_in = (const float*)d_in[0];
  int ci = 1;
  auto nxt = [&]() { return (const float*)d_in[ci++]; };

  BlockP bp[2];
  for (int b = 0; b < 2; ++b) {
    BlockP& P = bp[b];
    P.ca_Wg=nxt(); P.ca_Wk=nxt(); P.ca_Wo=nxt(); P.ca_Wq=nxt(); P.ca_Wv=nxt();
    P.ca_bg=nxt(); P.ca_bo=nxt(); P.ca_lnb=nxt(); P.ca_lng=nxt();
    P.mt_W1=nxt(); P.mt_W2=nxt(); P.mt_b1=nxt(); P.mt_b2=nxt(); P.mt_lnb=nxt(); P.mt_lng=nxt();
    P.op_Wa=nxt(); P.op_Wb=nxt(); P.op_Wo=nxt(); P.op_ba=nxt(); P.op_bb=nxt();
    P.op_bo=nxt(); P.op_lnb=nxt(); P.op_lng=nxt();
    P.pt_W1=nxt(); P.pt_W2=nxt(); P.pt_b1=nxt(); P.pt_b2=nxt(); P.pt_lnb=nxt(); P.pt_lng=nxt();
    P.ra_Wb2=nxt(); P.ra_Wg=nxt(); P.ra_Wk=nxt(); P.ra_Wo=nxt(); P.ra_Wq=nxt(); P.ra_Wv=nxt();
    P.ra_bg=nxt(); P.ra_bo=nxt(); P.ra_lnb=nxt(); P.ra_lng=nxt(); P.ra_lnzb=nxt(); P.ra_lnzg=nxt();
    P.te_Wb2=nxt(); P.te_Wg=nxt(); P.te_Wk=nxt(); P.te_Wo=nxt(); P.te_Wq=nxt(); P.te_Wv=nxt();
    P.te_bg=nxt(); P.te_bo=nxt(); P.te_lnb=nxt(); P.te_lng=nxt();
    P.ts_Wb2=nxt(); P.ts_Wg=nxt(); P.ts_Wk=nxt(); P.ts_Wo=nxt(); P.ts_Wq=nxt(); P.ts_Wv=nxt();
    P.ts_bg=nxt(); P.ts_bo=nxt(); P.ts_lnb=nxt(); P.ts_lng=nxt();
    P.ti_Wa=nxt(); P.ti_Wag=nxt(); P.ti_Wb=nxt(); P.ti_Wbg=nxt(); P.ti_Wg=nxt(); P.ti_Wo=nxt();
    P.ti_ba=nxt(); P.ti_bag=nxt(); P.ti_bb=nxt(); P.ti_bbg=nxt(); P.ti_bg=nxt(); P.ti_bo=nxt();
    P.ti_ln2b=nxt(); P.ti_ln2g=nxt(); P.ti_lnb=nxt(); P.ti_lng=nxt();
    P.to_Wa=nxt(); P.to_Wag=nxt(); P.to_Wb=nxt(); P.to_Wbg=nxt(); P.to_Wg=nxt(); P.to_Wo=nxt();
    P.to_ba=nxt(); P.to_bag=nxt(); P.to_bb=nxt(); P.to_bbg=nxt(); P.to_bg=nxt(); P.to_bo=nxt();
    P.to_ln2b=nxt(); P.to_ln2g=nxt(); P.to_lnb=nxt(); P.to_lng=nxt();
  }
  const float* coord_W = nxt(); const float* coord_b = nxt();
  const float* eW1=nxt(); const float* eW2=nxt(); const float* eb1=nxt(); const float* eb2=nxt();
  const float* nW1=nxt(); const float* nW2=nxt(); const float* nb1=nxt(); const float* nb2=nxt();
  const float* lin_W = nxt(); const float* lin_b = nxt();
  const float* z_in = (const float*)d_in[n_in - 1];

  // ---------------- outputs / workspace -------------------------------------
  float* mO = (float*)d_out;
  float* zO = mO + M_ELT;
  float* sO = zO + Z_ELT;

  float* w     = (float*)d_ws;
  float* pMLN  = w;
  float* pZLN  = pMLN + M_ELT;
  float* pQ    = pZLN + Z_ELT;
  float* pK    = pQ   + Z_ELT;
  float* pV    = pK   + Z_ELT;
  float* pG    = pV   + Z_ELT;
  float* pBIAS = pG   + Z_ELT;            // 294,912
  float* pHID  = pBIAS + 294912;          // 18,874,368
  float* pLOG  = pHID + 18874368;         // 18,874,368

  hipMemcpyAsync(mO, m_in, M_ELT * sizeof(float), hipMemcpyDeviceToDevice, stream);
  hipMemcpyAsync(zO, z_in, Z_ELT * sizeof(float), hipMemcpyDeviceToDevice, stream);

  // ---------------- launch helpers ------------------------------------------
  auto GEMM = [&](const float* A, long long offA, int rsA, int csA, long long bsA1, long long bsA2,
                  const float* B, long long offB, int rsB, int csB, long long bsB1, long long bsB2,
                  const float* bias, float* O, long long offO, int rsO, int csO,
                  long long bsO1, long long bsO2,
                  int M, int N, int K, int nb, int bDiv, float scale, int act, int accum) {
    const int tilesM = (M + 31) / 32;
    dim3 g((unsigned)((N + 31) / 32), (unsigned)((tilesM + 3) / 4), (unsigned)nb);
    k_gemm<<<g, dim3(128), 0, stream>>>(A, offA, rsA, csA, bsA1, bsA2,
                                        B, offB, rsB, csB, bsB1, bsB2, bias,
                                        O, offO, rsO, csO, bsO1, bsO2,
                                        M, N, K, bDiv, scale, act, accum);
  };
  auto GEMM1 = [&](const float* A, int rsA, const float* B, int rsB, const float* bias,
                   float* O, int rsO, int M, int N, int K, float scale, int act, int accum) {
    GEMM(A, 0, rsA, 1, 0, 0, B, 0, rsB, 1, 0, 0, bias, O, 0, rsO, 1, 0, 0,
         M, N, K, 1, 1, scale, act, accum);
  };
  auto LN = [&](const float* x, const float* g_, const float* b_, float* o_, int rows, int C) {
    k_layernorm<<<dim3((unsigned)((rows + 7) / 8)), dim3(256), 0, stream>>>(x, g_, b_, o_, rows, C);
  };
  auto SM = [&](float* x, long long rows, int L) {
    long long th = rows * 32;
    k_softmax<<<dim3((unsigned)((th + 255) / 256)), dim3(256), 0, stream>>>(x, rows, L);
  };
  auto MUL = [&](float* o_, const float* a_, const float* b_, long long n) {
    k_mul<<<dim3((unsigned)((n + 255) / 256)), dim3(256), 0, stream>>>(o_, a_, b_, n);
  };
  auto FMA2 = [&](float* d_, const float* a_, const float* b_, long long n) {
    k_fma<<<dim3((unsigned)((n + 255) / 256)), dim3(256), 0, stream>>>(d_, a_, b_, n);
  };

  // triangle multiplication: z += sigmoid(zln@Wg+bg) * (LN2(t) @ Wo + bo)
  auto TRIMUL = [&](const float* Wa, const float* Wag, const float* Wb, const float* Wbg,
                    const float* Wg_, const float* Wo_, const float* ba, const float* bag,
                    const float* bb, const float* bbg, const float* bg_, const float* bo_,
                    const float* ln2b, const float* ln2g, const float* lnb, const float* lng,
                    bool outgoing) {
    LN(zO, lng, lnb, pZLN, 36864, 128);
    GEMM1(pZLN, 128, Wag, 128, bag, pLOG, 128, 36864, 128, 128, 1.f, 2, 0);
    GEMM1(pZLN, 128, Wa,  128, ba,  pQ,   128, 36864, 128, 128, 1.f, 0, 0);
    MUL(pQ, pLOG, pQ, Z_ELT);
    GEMM1(pZLN, 128, Wbg, 128, bbg, pLOG, 128, 36864, 128, 128, 1.f, 2, 0);
    GEMM1(pZLN, 128, Wb,  128, bb,  pK,   128, 36864, 128, 128, 1.f, 0, 0);
    MUL(pK, pLOG, pK, Z_ELT);
    if (outgoing)
      GEMM(pQ, 0, 24576, 128, 0, 1, pK, 0, 128, 24576, 0, 1, nullptr,
           pV, 0, 24576, 128, 0, 1, 192, 192, 192, 128, 1, 1.f, 0, 0);
    else
      GEMM(pQ, 0, 128, 24576, 0, 1, pK, 0, 24576, 128, 0, 1, nullptr,
           pV, 0, 24576, 128, 0, 1, 192, 192, 192, 128, 1, 1.f, 0, 0);
    LN(pV, ln2g, ln2b, pG, 36864, 128);
    GEMM1(pZLN, 128, Wg_, 128, bg_, pLOG, 128, 36864, 128, 128, 1.f, 2, 0);
    GEMM1(pG, 128, Wo_, 128, bo_, pHID, 128, 36864, 128, 128, 1.f, 0, 0);
    FMA2(zO, pLOG, pHID, Z_ELT);
  };

  // triangle attention (optionally on transposed z, result added transposed)
  auto TRIATT = [&](const float* Wb2, const float* Wg_, const float* Wk_, const float* Wo_,
                    const float* Wq_, const float* Wv_, const float* bg_, const float* bo_,
                    const float* lnb, const float* lng, bool transposed) {
    const float* src = zO;
    float* oBuf = pHID;
    if (transposed) {
      k_transpose<<<dim3((unsigned)((Z_ELT + 255) / 256)), dim3(256), 0, stream>>>(pHID, zO, R, 128);
      src = pHID;
      oBuf = pHID + Z_ELT;
    }
    LN(src, lng, lnb, pZLN, 36864, 128);
    GEMM1(pZLN, 128, Wq_, 128, nullptr, pQ, 128, 36864, 128, 128, 1.f, 0, 0);
    GEMM1(pZLN, 128, Wk_, 128, nullptr, pK, 128, 36864, 128, 128, 1.f, 0, 0);
    GEMM1(pZLN, 128, Wv_, 128, nullptr, pV, 128, 36864, 128, 128, 1.f, 0, 0);
    GEMM1(pZLN, 128, Wg_, 128, bg_,     pG, 128, 36864, 128, 128, 1.f, 2, 0);
    GEMM1(pZLN, 128, Wb2, 4,   nullptr, pBIAS, 4, 36864, 4, 128, 1.f, 0, 0);
    for (int h = 0; h < 4; ++h) {
      GEMM(pQ, h * 32, 128, 1, 0, 24576, pK, h * 32, 1, 128, 0, 24576, nullptr,
           pLOG, 0, 192, 1, 0, 36864, 192, 192, 32, 192, 1, rs32, 0, 0);
      k_addbias_tri<<<dim3((unsigned)((7077888 + 255) / 256)), dim3(256), 0, stream>>>(
          pLOG, pBIAS, h, 4, R);
      SM(pLOG, 36864, 192);
      GEMM(pLOG, 0, 192, 1, 0, 36864, pV, h * 32, 128, 1, 0, 24576, nullptr,
           oBuf, h * 32, 128, 1, 0, 24576, 192, 32, 192, 192, 1, 1.f, 0, 0);
    }
    MUL(oBuf, oBuf, pG, Z_ELT);
    if (!transposed) {
      GEMM1(oBuf, 128, Wo_, 128, bo_, zO, 128, 36864, 128, 128, 1.f, 0, 1);
    } else {
      float* Ot = pHID + 2 * Z_ELT;
      GEMM1(oBuf, 128, Wo_, 128, bo_, Ot, 128, 36864, 128, 128, 1.f, 0, 0);
      k_add_t<<<dim3((unsigned)((Z_ELT + 255) / 256)), dim3(256), 0, stream>>>(zO, Ot, R, 128);
    }
  };

  // ---------------- Evoformer blocks ----------------------------------------
  for (int b = 0; b < 2; ++b) {
    const BlockP& P = bp[b];

    // ===== m += msa_row_att(m, z) =====
    LN(mO, P.ra_lng, P.ra_lnb, pMLN, 12288, 256);
    LN(zO, P.ra_lnzg, P.ra_lnzb, pZLN, 36864, 128);
    GEMM1(pMLN, 256, P.ra_Wq, 256, nullptr,  pQ, 256, 12288, 256, 256, 1.f, 0, 0);
    GEMM1(pMLN, 256, P.ra_Wk, 256, nullptr,  pK, 256, 12288, 256, 256, 1.f, 0, 0);
    GEMM1(pMLN, 256, P.ra_Wv, 256, nullptr,  pV, 256, 12288, 256, 256, 1.f, 0, 0);
    GEMM1(pMLN, 256, P.ra_Wg, 256, P.ra_bg,  pG, 256, 12288, 256, 256, 1.f, 2, 0);
    GEMM1(pZLN, 128, P.ra_Wb2, 8, nullptr, pBIAS, 8, 36864, 8, 128, 1.f, 0, 0);
    GEMM(pQ, 0, 256, 1, 32, 49152, pK, 0, 1, 256, 32, 49152, nullptr,
         pLOG, 0, 192, 1, 36864, 294912, 192, 192, 32, 512, 8, rs32, 0, 0);
    k_addbias_row<<<dim3((unsigned)((18874368 + 255) / 256)), dim3(256), 0, stream>>>(
        pLOG, pBIAS, S, 8, R);
    SM(pLOG, 98304, 192);
    GEMM(pLOG, 0, 192, 1, 36864, 294912, pV, 0, 256, 1, 32, 49152, nullptr,
         pHID, 0, 256, 1, 32, 49152, 192, 32, 192, 512, 8, 1.f, 0, 0);
    MUL(pHID, pHID, pG, M_ELT);
    GEMM1(pHID, 256, P.ra_Wo, 256, P.ra_bo, mO, 256, 12288, 256, 256, 1.f, 0, 1);

    // ===== m += msa_col_att(m) =====
    LN(mO, P.ca_lng, P.ca_lnb, pMLN, 12288, 256);
    GEMM1(pMLN, 256, P.ca_Wq, 256, nullptr, pQ, 256, 12288, 256, 256, 1.f, 0, 0);
    GEMM1(pMLN, 256, P.ca_Wk, 256, nullptr, pK, 256, 12288, 256, 256, 1.f, 0, 0);
    GEMM1(pMLN, 256, P.ca_Wv, 256, nullptr, pV, 256, 12288, 256, 256, 1.f, 0, 0);
    GEMM1(pMLN, 256, P.ca_Wg, 256, P.ca_bg, pG, 256, 12288, 256, 256, 1.f, 2, 0);
    GEMM(pQ, 0, 49152, 1, 32, 256, pK, 0, 1, 49152, 32, 256, nullptr,
         pLOG, 0, 64, 1, 4096, 32768, 64, 64, 32, 1536, 8, rs32, 0, 0);
    SM(pLOG, 98304, 64);
    GEMM(pLOG, 0, 64, 1, 4096, 32768, pV, 0, 49152, 1, 32, 256, nullptr,
         pHID, 0, 49152, 1, 32, 256, 64, 32, 64, 1536, 8, 1.f, 0, 0);
    MUL(pHID, pHID, pG, M_ELT);
    GEMM1(pHID, 256, P.ca_Wo, 256, P.ca_bo, mO, 256, 12288, 256, 256, 1.f, 0, 1);

    // ===== m += transition(m) =====
    LN(mO, P.mt_lng, P.mt_lnb, pMLN, 12288, 256);
    GEMM1(pMLN, 256, P.mt_W1, 1024, P.mt_b1, pHID, 1024, 12288, 1024, 256, 1.f, 1, 0);
    GEMM1(pHID, 1024, P.mt_W2, 256, P.mt_b2, mO, 256, 12288, 256, 1024, 1.f, 0, 1);

    // ===== z += outer_product_mean(m) =====
    LN(mO, P.op_lng, P.op_lnb, pMLN, 12288, 256);
    GEMM1(pMLN, 256, P.op_Wa, 32, P.op_ba, pQ, 32, 12288, 32, 256, 1.f, 0, 0);
    GEMM1(pMLN, 256, P.op_Wb, 32, P.op_bb, pK, 32, 12288, 32, 256, 1.f, 0, 0);
    k_opm<<<dim3(192, 192), dim3(128), 0, stream>>>(pQ, pK, P.op_Wo, P.op_bo, zO, S, R);

    // ===== pair stack =====
    TRIMUL(P.to_Wa, P.to_Wag, P.to_Wb, P.to_Wbg, P.to_Wg, P.to_Wo, P.to_ba, P.to_bag,
           P.to_bb, P.to_bbg, P.to_bg, P.to_bo, P.to_ln2b, P.to_ln2g, P.to_lnb, P.to_lng, true);
    TRIMUL(P.ti_Wa, P.ti_Wag, P.ti_Wb, P.ti_Wbg, P.ti_Wg, P.ti_Wo, P.ti_ba, P.ti_bag,
           P.ti_bb, P.ti_bbg, P.ti_bg, P.ti_bo, P.ti_ln2b, P.ti_ln2g, P.ti_lnb, P.ti_lng, false);
    TRIATT(P.ts_Wb2, P.ts_Wg, P.ts_Wk, P.ts_Wo, P.ts_Wq, P.ts_Wv, P.ts_bg, P.ts_bo,
           P.ts_lnb, P.ts_lng, false);
    TRIATT(P.te_Wb2, P.te_Wg, P.te_Wk, P.te_Wo, P.te_Wq, P.te_Wv, P.te_bg, P.te_bo,
           P.te_lnb, P.te_lng, true);
    LN(zO, P.pt_lng, P.pt_lnb, pZLN, 36864, 128);
    GEMM1(pZLN, 128, P.pt_W1, 512, P.pt_b1, pHID, 512, 36864, 512, 128, 1.f, 1, 0);
    GEMM1(pHID, 512, P.pt_W2, 128, P.pt_b2, zO, 128, 36864, 128, 512, 1.f, 0, 1);
  }

  // ---------------- head: s = egnn(m[:,0] @ lin, coords) --------------------
  float* pSL  = pQ;
  float* pCO  = pK;
  float* pEIN = pV;
  float* pEF1 = pG;
  float* pEF  = pHID;
  float* pAGG = pBIAS;

  GEMM1(mO, 256, lin_W, 384, lin_b, pSL, 384, 192, 384, 256, 1.f, 0, 0);
  GEMM1(pSL, 384, coord_W, 3, coord_b, pCO, 3, 192, 3, 384, 1.f, 0, 0);
  {
    long long n = 382LL * 769;
    k_egnn_edges<<<dim3((unsigned)((n + 255) / 256)), dim3(256), 0, stream>>>(pSL, pCO, pEIN, 192);
  }
  GEMM1(pEIN, 769, eW1, 384, eb1, pEF1, 384, 382, 384, 769, 1.f, 3, 0);
  GEMM1(pEF1, 384, eW2, 384, eb2, pEF,  384, 382, 384, 384, 1.f, 0, 0);
  {
    long long n = 192LL * 384;
    k_egnn_agg<<<dim3((unsigned)((n + 255) / 256)), dim3(256), 0, stream>>>(pEF, pAGG, 192);
    long long n2 = 192LL * 768;
    k_concat2<<<dim3((unsigned)((n2 + 255) / 256)), dim3(256), 0, stream>>>(pEIN, pSL, pAGG, 192, 384);
  }
  GEMM1(pEIN, 768, nW1, 384, nb1, pEF1, 384, 192, 384, 768, 1.f, 3, 0);
  GEMM1(pEF1, 384, nW2, 384, nb2, sO,   384, 192, 384, 384, 1.f, 0, 0);
}
